// TKANCell_12919261626543
// MI455X (gfx1250) — compile-verified, matrix operand order checked by reading, not run on hardware
//
#include <hip/hip_runtime.h>
#include <hip/hip_bf16.h>
#include <math.h>

// ---------------- problem constants ----------------
#define B_   64
#define T_   2048
#define D_   128
#define U_   128
#define NS   3
#define G3   384      // 3*U

typedef __attribute__((ext_vector_type(16))) __bf16 v16bf;
typedef __attribute__((ext_vector_type(2)))  __bf16 v2bf;
typedef __attribute__((ext_vector_type(8)))  float  v8f;

union BfVec { v16bf v; unsigned int u[8]; };
union FVec  { v8f  v; float f[8]; };

// ---------------- helpers ----------------
// f32 -> bf16 via hardware convert (backend emits v_cvt_pk_bf16_f32 / v_cvt_bf16_f32)
__device__ __forceinline__ unsigned int pack2(float a, float b) {
  v2bf p;
  p.x = (__bf16)a;
  p.y = (__bf16)b;
  return __builtin_bit_cast(unsigned int, p);
}
__device__ __forceinline__ unsigned short f2bf(float f) {
  return __builtin_bit_cast(unsigned short, (__bf16)f);
}
__device__ __forceinline__ float bf2f(unsigned int h) {
  unsigned int u = (h & 0xFFFFu) << 16;
  return __builtin_bit_cast(float, u);
}
// ISA 7.12.2: 16-bit A/B matrix K-offset for VGPR v, lane-half
__device__ __forceinline__ int kofs(int v, int half) {
  return (v & 3) * 2 + (v >> 2) * 16 + half * 8;
}
__device__ __forceinline__ float sigf(float x) { return 1.0f / (1.0f + __expf(-x)); }

__device__ __forceinline__ v8f wmma_bf16(const BfVec& a, const BfVec& b, v8f c) {
  return __builtin_amdgcn_wmma_f32_16x16x32_bf16(false, a.v, false, b.v,
                                                 (short)0, c, false, false);
}

// pack A-tile from f32 rows (global x or LDS s): row = base of row (stride 1 in k)
__device__ __forceinline__ void packA_f32(const float* __restrict__ row, int kbase,
                                          int half, BfVec& a) {
#pragma unroll
  for (int v = 0; v < 8; ++v) {
    int k = kbase + kofs(v, half);
    const float2 f2 = *(const float2*)(row + k);   // k is even -> 8B aligned
    a.u[v] = pack2(f2.x, f2.y);
  }
}
// pack A-tile from bf16 row (LDS); k even & row base dword-aligned by construction
__device__ __forceinline__ void packA_bf16(const unsigned short* __restrict__ row,
                                           int kbase, int half, BfVec& a) {
  const unsigned int* r32 = (const unsigned int*)row;
#pragma unroll
  for (int v = 0; v < 8; ++v) {
    int k = kbase + kofs(v, half);
    a.u[v] = r32[k >> 1];
  }
}
// load pre-packed B tile: 8 contiguous dwords per lane -> 2x global_load_b128
__device__ __forceinline__ void loadB(const unsigned int* __restrict__ wsb, int tile,
                                      int lane, BfVec& b) {
  const uint4* p = (const uint4*)(wsb + ((size_t)tile * 32 + lane) * 8);
  uint4 q0 = p[0], q1 = p[1];
  b.u[0] = q0.x; b.u[1] = q0.y; b.u[2] = q0.z; b.u[3] = q0.w;
  b.u[4] = q1.x; b.u[5] = q1.y; b.u[6] = q1.z; b.u[7] = q1.w;
}
// load C (accumulator init) from bf16-pair D-layout store (4 dwords / lane)
__device__ __forceinline__ void loadC_bf16(const unsigned int* __restrict__ p, FVec& c) {
  uint4 q = *(const uint4*)p;
  unsigned int uu[4] = { q.x, q.y, q.z, q.w };
#pragma unroll
  for (int i = 0; i < 4; ++i) {
    c.f[2 * i]     = bf2f(uu[i]);
    c.f[2 * i + 1] = bf2f(uu[i] >> 16);
  }
}

// ================= phase 0: pack f32 weights [mat][K][N] -> WMMA B layout ==========
// dst dword index = (((mat*Nt + nt)*Kt + kt)*32 + lane)*8 + v  (bf16 pair per dword)
__global__ void pack_b_kernel(const float* __restrict__ src, unsigned int* __restrict__ dst,
                              int nmat, int K, int N) {
  int Kt = K >> 5, Nt = N >> 4;
  int total = nmat * Nt * Kt * 256;
  int idx = blockIdx.x * blockDim.x + threadIdx.x;
  if (idx >= total) return;
  int v    = idx & 7;
  int lane = (idx >> 3) & 31;
  int rest = idx >> 8;
  int kt   = rest % Kt;
  int nt   = (rest / Kt) % Nt;
  int mat  = rest / (Kt * Nt);
  int half = lane >> 4, n = lane & 15;
  int k = kt * 32 + kofs(v, half);
  const float* w = src + (size_t)mat * K * N;
  dst[idx] = pack2(w[(size_t)k * N + nt * 16 + n],
                   w[(size_t)(k + 1) * N + nt * 16 + n]);
}

// ================= phase 1: hoisted x-projections (parallel over T) ================
// xg[t]   = x[:,t,:] @ kernel + bias            (64 x 384)  -> bf16, D-tile layout
// xsub[t] = x[:,t,:] @ sub_kernel_x[n]  n=0..2  (3 x 64x128) -> bf16, D-tile layout
// D-tile layout: dword idx = ((t*96 + job)*32 + lane)*4 + i
__global__ void __launch_bounds__(256)
phase1_xproj(const float* __restrict__ x, const float* __restrict__ bias,
             const unsigned int* __restrict__ kernelB,
             const unsigned int* __restrict__ subxB,
             unsigned int* __restrict__ xg, unsigned int* __restrict__ xsub) {
  int t = blockIdx.x;
  int tid = threadIdx.x, w = tid >> 5, lane = tid & 31, half = lane >> 4, l16 = lane & 15;

  for (int j = w; j < 192; j += 8) {
    FVec acc; BfVec a, b;
    if (j < 96) {                       // --- xg job: tile (mt, nt), nt in [0,24)
      int mt = j / 24, nt = j % 24;
      const float* row = x + ((size_t)(mt * 16 + l16) * T_ + t) * D_;
      float cb = bias[nt * 16 + l16];
#pragma unroll
      for (int r = 0; r < 8; ++r) acc.f[r] = cb;
#pragma unroll
      for (int kt = 0; kt < 4; ++kt) {
        packA_f32(row, kt * 32, half, a);
        loadB(kernelB, nt * 4 + kt, lane, b);
        acc.v = wmma_bf16(a, b, acc.v);
      }
      uint4 q = { pack2(acc.f[0], acc.f[1]), pack2(acc.f[2], acc.f[3]),
                  pack2(acc.f[4], acc.f[5]), pack2(acc.f[6], acc.f[7]) };
      *(uint4*)(xg + (((size_t)t * 96 + j) * 32 + lane) * 4) = q;
    } else {                            // --- xsub job: (n, mt, nt), nt in [0,8)
      int j2 = j - 96;
      int n = j2 >> 5, mt = (j2 >> 3) & 3, nt = j2 & 7;
      const float* row = x + ((size_t)(mt * 16 + l16) * T_ + t) * D_;
#pragma unroll
      for (int r = 0; r < 8; ++r) acc.f[r] = 0.0f;
#pragma unroll
      for (int kt = 0; kt < 4; ++kt) {
        packA_f32(row, kt * 32, half, a);
        loadB(subxB, (n * 8 + nt) * 4 + kt, lane, b);
        acc.v = wmma_bf16(a, b, acc.v);
      }
      uint4 q = { pack2(acc.f[0], acc.f[1]), pack2(acc.f[2], acc.f[3]),
                  pack2(acc.f[4], acc.f[5]), pack2(acc.f[6], acc.f[7]) };
      *(uint4*)(xsub + (((size_t)t * 96 + j2) * 32 + lane) * 4) = q;
    }
  }
}

// ================= phase 2: sequential recurrence, one persistent workgroup ========
// 1024 threads = 32 waves on one WGP. LDS (dynamic, 208 KB):
//   sh_h  bf16 [64][128]      16 KB   (h_{t-1}, WMMA A operand of gates GEMM)
//   sh_s  f32  [3][64][128]   96 KB   (sub-layer state, f32 for the s-update)
//   sh_ai bf16 [3][64][128]   48 KB   (agg_in staging, A operand of dense GEMM)
//   sh_so bf16 [3][64][128]   48 KB   (sub_o; linear view == "faithful reshape" flat)
// c lives in 8 VGPRs per lane for the whole T loop (gate tile == output tile).
__global__ void __launch_bounds__(1024)
phase2_recurrence(const unsigned int* __restrict__ recB,
                  const unsigned int* __restrict__ subhB,
                  const unsigned int* __restrict__ denseB,
                  const unsigned int* __restrict__ aggB,
                  const unsigned int* __restrict__ xg,
                  const unsigned int* __restrict__ xsub,
                  const float* __restrict__ dense_b,
                  const float* __restrict__ agg_b,
                  const float* __restrict__ tkan,      // (3, 256): rh | rx
                  float* __restrict__ out) {
  extern __shared__ unsigned char smem[];
  unsigned short* sh_h  = (unsigned short*)(smem);
  float*          sh_s  = (float*)(smem + 16384);
  unsigned short* sh_ai = (unsigned short*)(smem + 16384 + 98304);
  unsigned short* sh_so = (unsigned short*)(smem + 16384 + 98304 + 49152);

  int tid = threadIdx.x, w = tid >> 5, lane = tid & 31, half = lane >> 4, l16 = lane & 15;
  int mtw = w & 3, ngw = w >> 2;      // this wave's (batch-tile, U-tile)

  for (int i = tid; i < 64 * 128; i += 1024) sh_h[i] = 0;
  for (int i = tid; i < NS * 64 * 128; i += 1024) sh_s[i] = 0.0f;
  float c[8];
#pragma unroll
  for (int r = 0; r < 8; ++r) c[r] = 0.0f;
  __syncthreads();

  for (int t = 0; t < T_; ++t) {
    // ---------- Stage G: gates = sigmoid(xg[t] + h@Wr); c = f*c + i*tanh(cg) ------
    {
      BfVec ah[4];
#pragma unroll
      for (int kt = 0; kt < 4; ++kt)
        packA_bf16(sh_h + (mtw * 16 + l16) * 128, kt * 32, half, ah[kt]);
      FVec g[3];
#pragma unroll
      for (int gi = 0; gi < 3; ++gi) {
        int nt = gi * 8 + ngw;
        loadC_bf16(xg + (((size_t)t * 96 + (mtw * 24 + nt)) * 32 + lane) * 4, g[gi]);
        BfVec b;
#pragma unroll
        for (int kt = 0; kt < 4; ++kt) {
          loadB(recB, nt * 4 + kt, lane, b);
          g[gi].v = wmma_bf16(ah[kt], b, g[gi].v);
        }
      }
#pragma unroll
      for (int r = 0; r < 8; ++r) {
        float ig = sigf(g[0].f[r]);
        float fg = sigf(g[1].f[r]);
        float cg = tanhf(sigf(g[2].f[r]));     // reference applies sigmoid then tanh
        c[r] = fg * c[r] + ig * cg;
      }
    }
    // ---------- Stage A: agg_in = xsub[t] + s@Wh  (96 tiles, 3 per wave) ----------
#pragma unroll
    for (int jj = 0; jj < 3; ++jj) {
      int ja = w + jj * 32;
      int n = ja >> 5, mt = (ja >> 3) & 3, nt = ja & 7;
      FVec acc; BfVec a, b;
      loadC_bf16(xsub + (((size_t)t * 96 + (n * 32 + mt * 8 + nt)) * 32 + lane) * 4, acc);
      const float* srow = sh_s + (n * 64 + mt * 16 + l16) * 128;
#pragma unroll
      for (int kt = 0; kt < 4; ++kt) {
        packA_f32(srow, kt * 32, half, a);
        loadB(subhB, (n * 8 + nt) * 4 + kt, lane, b);
        acc.v = wmma_bf16(a, b, acc.v);
      }
      int col = nt * 16 + l16;
#pragma unroll
      for (int r = 0; r < 8; ++r) {
        int m = mt * 16 + r + 8 * half;                       // C/D row (ISA 7.12.2)
        sh_ai[(n * 64 + m) * 128 + col] = f2bf(acc.f[r]);
      }
    }
    __syncthreads();
    // ---------- Stage D: sub_o = relu(agg_in@dense + db); s = rh*sub_o + rx*s -----
#pragma unroll
    for (int jj = 0; jj < 3; ++jj) {
      int jd = w + jj * 32;
      int n = jd >> 5, mt = (jd >> 3) & 3, nt = jd & 7;
      FVec acc; BfVec a, b;
      int col = nt * 16 + l16;
      float cb = dense_b[n * 128 + col];
#pragma unroll
      for (int r = 0; r < 8; ++r) acc.f[r] = cb;
#pragma unroll
      for (int kt = 0; kt < 4; ++kt) {
        packA_bf16(sh_ai + (n * 64 + mt * 16 + l16) * 128, kt * 32, half, a);
        loadB(denseB, (n * 8 + nt) * 4 + kt, lane, b);
        acc.v = wmma_bf16(a, b, acc.v);
      }
      float rh = tkan[n * 256 + col];
      float rx = tkan[n * 256 + 128 + col];
#pragma unroll
      for (int r = 0; r < 8; ++r) {
        float so = fmaxf(acc.f[r], 0.0f);
        int idx = (n * 64 + mt * 16 + r + 8 * half) * 128 + col;
        sh_so[idx] = f2bf(so);
        sh_s[idx]  = rh * so + rx * sh_s[idx];
      }
    }
    __syncthreads();
    // ---------- Stage O: o = sigmoid(flat@agg + ab); h = o*tanh(c) ----------------
    // flat(64x384) == linear view of sh_so (the "faithful" TF reshape is free here)
    {
      FVec acc; BfVec a, b;
      int col = ngw * 16 + l16;
      float cb = agg_b[col];
#pragma unroll
      for (int r = 0; r < 8; ++r) acc.f[r] = cb;
#pragma unroll
      for (int kt = 0; kt < 12; ++kt) {
        packA_bf16(sh_so + (mtw * 16 + l16) * 384, kt * 32, half, a);
        loadB(aggB, ngw * 12 + kt, lane, b);
        acc.v = wmma_bf16(a, b, acc.v);
      }
#pragma unroll
      for (int r = 0; r < 8; ++r) {
        int bb = mtw * 16 + r + 8 * half;
        float h = sigf(acc.f[r]) * tanhf(c[r]);   // same lane owns c for this (b,u)
        sh_h[bb * 128 + col] = f2bf(h);
        out[((size_t)bb * T_ + t) * U_ + col] = h;
      }
    }
    __syncthreads();
  }
}

// ================= host side ================
// workspace layout (bytes)
#define WSB_TILE_BYTES   (96 * 32 * 8 * 4)       // 98304 per packed matrix family
#define WS_KERNEL_B      0
#define WS_SUBX_B        (WSB_TILE_BYTES * 1)
#define WS_REC_B         (WSB_TILE_BYTES * 2)
#define WS_SUBH_B        (WSB_TILE_BYTES * 3)
#define WS_DENSE_B       (WSB_TILE_BYTES * 4)
#define WS_AGG_B         (WSB_TILE_BYTES * 5)
#define WS_XG            ((size_t)1 << 20)
#define XG_BYTES         ((size_t)T_ * 96 * 32 * 8 * 2)   // 96 MB bf16
#define WS_XSUB          (WS_XG + XG_BYTES)
#define SMEM_BYTES       (16384 + 98304 + 49152 + 49152)  // 208 KB < 320 KB WGP LDS

extern "C" void kernel_launch(void* const* d_in, const int* in_sizes, int n_in,
                              void* d_out, int out_size, void* d_ws, size_t ws_size,
                              hipStream_t stream) {
  const float* x      = (const float*)d_in[0];
  const float* kern   = (const float*)d_in[1];
  const float* rec    = (const float*)d_in[2];
  const float* bias   = (const float*)d_in[3];
  const float* subx   = (const float*)d_in[4];
  const float* subh   = (const float*)d_in[5];
  const float* tkan   = (const float*)d_in[6];
  const float* dw     = (const float*)d_in[7];
  const float* db     = (const float*)d_in[8];
  const float* aw     = (const float*)d_in[9];
  const float* ab     = (const float*)d_in[10];
  float* out          = (float*)d_out;
  unsigned char* ws   = (unsigned char*)d_ws;

  unsigned int* kernelB = (unsigned int*)(ws + WS_KERNEL_B);
  unsigned int* subxB   = (unsigned int*)(ws + WS_SUBX_B);
  unsigned int* recB    = (unsigned int*)(ws + WS_REC_B);
  unsigned int* subhB   = (unsigned int*)(ws + WS_SUBH_B);
  unsigned int* denseB  = (unsigned int*)(ws + WS_DENSE_B);
  unsigned int* aggB    = (unsigned int*)(ws + WS_AGG_B);
  unsigned int* xg      = (unsigned int*)(ws + WS_XG);
  unsigned int* xsub    = (unsigned int*)(ws + WS_XSUB);

  // phase 0: pack all weight matrices into WMMA B-register layout (bf16)
  {
    int tot;
    tot = 1 * 24 * 4 * 256;  // kernel 128x384
    pack_b_kernel<<<(tot + 255) / 256, 256, 0, stream>>>(kern, kernelB, 1, 128, 384);
    tot = 3 * 8 * 4 * 256;   // sub_kernel_x 3x128x128
    pack_b_kernel<<<(tot + 255) / 256, 256, 0, stream>>>(subx, subxB, 3, 128, 128);
    tot = 1 * 24 * 4 * 256;  // recurrent_kernel 128x384
    pack_b_kernel<<<(tot + 255) / 256, 256, 0, stream>>>(rec, recB, 1, 128, 384);
    tot = 3 * 8 * 4 * 256;   // sub_kernel_h 3x128x128
    pack_b_kernel<<<(tot + 255) / 256, 256, 0, stream>>>(subh, subhB, 3, 128, 128);
    tot = 3 * 8 * 4 * 256;   // dense_w 3x128x128
    pack_b_kernel<<<(tot + 255) / 256, 256, 0, stream>>>(dw, denseB, 3, 128, 128);
    tot = 1 * 8 * 12 * 256;  // agg_w 384x128
    pack_b_kernel<<<(tot + 255) / 256, 256, 0, stream>>>(aw, aggB, 1, 384, 128);
  }

  // phase 1: hoisted x projections, parallel over all T (bandwidth phase)
  phase1_xproj<<<T_, 256, 0, stream>>>(x, bias, kernelB, subxB, xg, xsub);

  // phase 2: sequential recurrence, one persistent workgroup (latency phase)
  phase2_recurrence<<<1, 1024, SMEM_BYTES, stream>>>(recB, subhB, denseB, aggB,
                                                     xg, xsub, db, ab, tkan, out);
}
